// BoundaryAwareDownConv_50148038148454
// MI455X (gfx1250) — matrix operand order repaired and checked by reading, not compiled, not run on hardware
//
#include <hip/hip_runtime.h>

// Problem constants (match reference)
#define DM      512
#define T_LEN   8192
#define BATCH   8
#define STRIDE_ 8
#define WORDS   (T_LEN / STRIDE_)      // 1024
#define ROWS    (BATCH * WORDS)        // 8192
#define LN_EPS  1e-5f

typedef _Float16 v4h  __attribute__((ext_vector_type(4)));
typedef _Float16 v8h  __attribute__((ext_vector_type(8)));
typedef _Float16 v16h __attribute__((ext_vector_type(16)));
typedef float    v8f  __attribute__((ext_vector_type(8)));

// ---------------------------------------------------------------------------
// Kernel 0: one-shot f32 -> f16 conversion of w_proj (1 MB read, trivial)
// ---------------------------------------------------------------------------
__global__ __launch_bounds__(256)
void cvt_kernel(const float* __restrict__ wp, _Float16* __restrict__ wp16) {
    const int i = blockIdx.x * 256 + threadIdx.x;   // DM*DM / 2 threads, float2 each
    float2 v = reinterpret_cast<const float2*>(wp)[i];
    v4h o; o.x = (_Float16)v.x; o.y = (_Float16)v.y;
    *reinterpret_cast<_Float16*>(&o);               // silence unused halves
    _Float16* dst = wp16 + i * 2;
    dst[0] = (_Float16)v.x; dst[1] = (_Float16)v.y;
}

// ---------------------------------------------------------------------------
// Kernel 1: segment-mean pooling, f32 in -> f16 out (GEMM A operand).
// One block per (batch, word) row; 128 threads x float4 -> coalesced 128 MB read
// (the HBM-bound part).  input_ids uniform per block -> scalar loads.
// ---------------------------------------------------------------------------
__global__ __launch_bounds__(128)
void pool_kernel(const float* __restrict__ x,
                 const int* __restrict__ ids,
                 _Float16* __restrict__ pooled16) {
    const int row = blockIdx.x;            // b * WORDS + w
    const int b   = row >> 10;
    const int w   = row & (WORDS - 1);
    const int t0  = b * T_LEN + w * STRIDE_;
    const int c   = threadIdx.x;           // float4 chunk 0..127

    const float4* x4 = reinterpret_cast<const float4*>(x);
    float4 s = make_float4(0.f, 0.f, 0.f, 0.f);
    float  cnt = 0.f;
#pragma unroll
    for (int j = 0; j < STRIDE_; ++j) {
        const int tok = t0 + j;
        if (ids[tok] != 0) {               // SPACE_ID == 0 excluded
            float4 v = x4[tok * (DM / 4) + c];
            s.x += v.x; s.y += v.y; s.z += v.z; s.w += v.w;
            cnt += 1.f;
        }
    }
    const float inv = 1.f / fmaxf(cnt, 1.f);
    v4h o;
    o.x = (_Float16)(s.x * inv); o.y = (_Float16)(s.y * inv);
    o.z = (_Float16)(s.z * inv); o.w = (_Float16)(s.w * inv);
    *reinterpret_cast<v4h*>(&pooled16[row * DM + c * 4]) = o;   // 8B store
}

// ---------------------------------------------------------------------------
// Kernel 2: fused  proj = pooled @ W^T + b  ->  LayerNorm(proj)
// Block = 256 threads (8 wave32).  Block owns a 32-row M-tile x all 512 cols.
// K in 32-wide panels staged to LDS as f16; V_WMMA_F32_16X16X32_F16 accumulate.
//
// f16 fragment layouts used here (contiguous k order, same for A and B):
//   A lane L (M = L%16, h = L/16): elements e=0..7  -> K = e + 16h
//                                  elements e=8..15 -> K = 8 + (e-8) + 16h
//     => two 8-half (16 B) chunks at K-offsets 16h and 16h+8.
//   B lane L (N = L%16, h = L/16): element e -> K = e + 16h
//     => one contiguous 16-half (32 B) chunk at K-offset 16h.
//   C/D f32: VGPR r, half h -> (M = r + 8h, N = L%16).
// ---------------------------------------------------------------------------
#define KP    32                 // K panel width (one WMMA k-step)
#define LDH   (KP + 8)           // padded LDS row stride in halves (40 -> 80 B,
                                 // 20-bank lane stride, gcd(20,64)=4: conflict-free)
#define MT    32                 // M rows per block (2 m-tiles per wave)
#define LDP   (DM + 4)           // padded f32 proj-tile row stride (516)

#define AS_HALVES (MT * LDH)               // 1280
#define BS_HALVES (DM * LDH)               // 20480
#define SMEM_BYTES ((AS_HALVES + BS_HALVES) * 2)   // 43,520 B (P needs 33,024)

__global__ __launch_bounds__(256)
void gemm_ln_kernel(const _Float16* __restrict__ pooled16,
                    const _Float16* __restrict__ wp16,   // [DM][DM] row n = w_proj[n,:]
                    const float* __restrict__ bias,
                    const float* __restrict__ gamma,
                    const float* __restrict__ beta,
                    float* __restrict__ out) {
    __shared__ __align__(16) unsigned char smem_raw[SMEM_BYTES];
    _Float16* As = reinterpret_cast<_Float16*>(smem_raw);
    _Float16* Bs = As + AS_HALVES;
    float*    P  = reinterpret_cast<float*>(smem_raw);   // aliases As/Bs post-GEMM

    const int tid  = threadIdx.x;
    const int wave = tid >> 5;             // 0..7
    const int lane = tid & 31;
    const int lo   = lane & 15;
    const int hi   = lane >> 4;
    const int row0 = blockIdx.x * MT;
    const int nbase = wave * 64;           // wave covers cols [nbase, nbase+64)

    v8f acc[2][4];
#pragma unroll
    for (int mt = 0; mt < 2; ++mt)
#pragma unroll
        for (int j = 0; j < 4; ++j) acc[mt][j] = (v8f)(0.f);

    // ---- K loop: 16 panels of 32 ----
    for (int p = 0; p < DM / KP; ++p) {
        const int k0 = p * KP;

        // Stage As: 32 rows x 32 halves = 128 x (8-half/16 B) chunks, threads 0..127
        if (tid < 128) {
            const int m = tid >> 2, c4 = tid & 3;
            v8h v = *reinterpret_cast<const v8h*>(&pooled16[(row0 + m) * DM + k0 + c4 * 8]);
            *reinterpret_cast<v8h*>(&As[m * LDH + c4 * 8]) = v;
        }
        // Stage Bs: 512 rows x 32 halves = 2048 chunks, 8 per thread
#pragma unroll
        for (int i = 0; i < 8; ++i) {
            const int idx = i * 256 + tid;
            const int n = idx >> 2, c4 = idx & 3;
            v8h v = *reinterpret_cast<const v8h*>(&wp16[n * DM + k0 + c4 * 8]);
            *reinterpret_cast<v8h*>(&Bs[n * LDH + c4 * 8]) = v;
            if (k0 + KP < DM)   // prefetch next panel (global_prefetch_b8)
                __builtin_prefetch(&wp16[n * DM + k0 + KP + c4 * 8], 0, 1);
        }
        __syncthreads();

        // A fragments for the two m-tiles (two 16 B LDS chunks each)
        v16h afrag[2];
#pragma unroll
        for (int mt = 0; mt < 2; ++mt) {
            const int m = mt * 16 + lo;
            v8h a0 = *reinterpret_cast<const v8h*>(&As[m * LDH + 16 * hi]);
            v8h a1 = *reinterpret_cast<const v8h*>(&As[m * LDH + 16 * hi + 8]);
#pragma unroll
            for (int e = 0; e < 8; ++e) { afrag[mt][e] = a0[e]; afrag[mt][e + 8] = a1[e]; }
        }
        // 4 N-tiles: one B fragment each, 2 WMMAs (one per m-tile)
#pragma unroll
        for (int j = 0; j < 4; ++j) {
            const int n = nbase + j * 16 + lo;
            v16h bfrag = *reinterpret_cast<const v16h*>(&Bs[n * LDH + 16 * hi]);
            acc[0][j] = __builtin_amdgcn_wmma_f32_16x16x32_f16(
                false, afrag[0], false, bfrag, (short)0, acc[0][j], false, false);
            acc[1][j] = __builtin_amdgcn_wmma_f32_16x16x32_f16(
                false, afrag[1], false, bfrag, (short)0, acc[1][j], false, false);
        }
        __syncthreads();   // panel consumed; safe to restage
    }

    // ---- Epilogue: two 16-row passes of (bias -> LDS spill -> LayerNorm) ----
#pragma unroll
    for (int mt = 0; mt < 2; ++mt) {
        // Spill proj (+bias) into P[16][LDP] (aliases staging LDS)
#pragma unroll
        for (int j = 0; j < 4; ++j) {
            const int n = nbase + j * 16 + lo;
            const float bb = bias[n];
#pragma unroll
            for (int r = 0; r < 8; ++r) {
                const int m = r + 8 * hi;          // C/D layout
                P[m * LDP + n] = acc[mt][j][r] + bb;
            }
        }
        __syncthreads();

        // Each wave LayerNorms rows {wave, wave+8} of this 16-row group
#pragma unroll
        for (int m = wave; m < 16; m += 8) {
            float s = 0.f, s2 = 0.f;
#pragma unroll
            for (int c = lane; c < DM; c += 32) {
                const float v = P[m * LDP + c];
                s += v; s2 += v * v;
            }
#pragma unroll
            for (int off = 16; off >= 1; off >>= 1) {
                s  += __shfl_xor(s,  off, 32);
                s2 += __shfl_xor(s2, off, 32);
            }
            const float mean = s * (1.f / DM);
            const float var  = s2 * (1.f / DM) - mean * mean;
            const float rstd = rsqrtf(var + LN_EPS);
#pragma unroll
            for (int c = lane; c < DM; c += 32) {
                const float v = P[m * LDP + c];
                out[(row0 + mt * 16 + m) * DM + c] = (v - mean) * rstd * gamma[c] + beta[c];
            }
        }
        __syncthreads();   // P reused by next 16-row group
    }
}

// ---------------------------------------------------------------------------
extern "C" void kernel_launch(void* const* d_in, const int* in_sizes, int n_in,
                              void* d_out, int out_size, void* d_ws, size_t ws_size,
                              hipStream_t stream) {
    const float* x     = (const float*)d_in[0];
    const int*   ids   = (const int*)  d_in[1];
    const float* wproj = (const float*)d_in[2];
    const float* bias  = (const float*)d_in[3];
    const float* gamma = (const float*)d_in[4];
    const float* beta  = (const float*)d_in[5];
    float*       out   = (float*)d_out;

    _Float16* pooled16 = (_Float16*)d_ws;            // ROWS*DM halves = 8 MB
    _Float16* wp16     = pooled16 + ROWS * DM;       // DM*DM halves   = 0.5 MB

    cvt_kernel <<<(DM * DM / 2) / 256, 256, 0, stream>>>(wproj, wp16);
    pool_kernel<<<ROWS, 128, 0, stream>>>(x, ids, pooled16);
    gemm_ln_kernel<<<ROWS / MT, 256, 0, stream>>>(pooled16, wp16, bias, gamma, beta, out);
}
